// EdgeGCNLayer_3453153706428
// MI455X (gfx1250) — matrix operand order repaired and checked by reading, not compiled, not run on hardware
//
#include <hip/hip_runtime.h>
#include <hip/hip_bf16.h>

#define N_NODES  100000
#define N_EDGES  1600000
#define IN_CH    128
#define OUT_CH   64
#define EDGE_DIM 32

typedef __attribute__((ext_vector_type(16))) __bf16 v16bf;
typedef __attribute__((ext_vector_type(2)))  __bf16 v2bf;
typedef __attribute__((ext_vector_type(8)))  float  v8f;

// f32 -> bf16 RNE, native packed convert (v_cvt_pk_bf16_f32 confirmed in codegen)
__device__ __forceinline__ v2bf pk2(float a, float b) {
#if __has_builtin(__builtin_amdgcn_cvt_pk_bf16_f32)
  return __builtin_amdgcn_cvt_pk_bf16_f32(a, b);
#else
  v2bf r; r[0] = (__bf16)a; r[1] = (__bf16)b; return r;
#endif
}
__device__ __forceinline__ __bf16 f2bf(float f) { return (__bf16)f; }

union FragU { v16bf v; v2bf p[8]; };

__device__ __forceinline__ void pack8(FragU& u, int base2, float4 lo, float4 hi4) {
  u.p[base2 + 0] = pk2(lo.x, lo.y);
  u.p[base2 + 1] = pk2(lo.z, lo.w);
  u.p[base2 + 2] = pk2(hi4.x, hi4.y);
  u.p[base2 + 3] = pk2(hi4.z, hi4.w);
}

__global__ void zero_f32_kernel(float* __restrict__ p, int n) {
  int i = blockIdx.x * blockDim.x + threadIdx.x;
  if (i < n) p[i] = 0.0f;
}

// Prepack WMMA-ready bf16 B-fragments for W_lin (4 ksteps x 4 ntiles) and
// W_edge (4 ntiles). B-frag layout: lane = col (lane&15), K = (lane<16?0:8)+{0..7,16..23}.
__global__ __launch_bounds__(640)
void pack_w_kernel(const float* __restrict__ Wl, const float* __restrict__ We,
                   v16bf* __restrict__ Wlf, v16bf* __restrict__ Wef) {
  const int t = threadIdx.x;
  if (t < 512) {
    const int lane = t & 31, ntile = (t >> 5) & 3, kk = t >> 7;
    const int ch = ntile * 16 + (lane & 15);
    const int kbase = kk * 32 + ((lane >> 4) ? 8 : 0);
    v16bf f;
#pragma unroll
    for (int j = 0; j < 16; ++j) f[j] = f2bf(Wl[(size_t)(kbase + j + ((j >> 3) << 3)) * OUT_CH + ch]);
    Wlf[t] = f;
  } else if (t < 640) {
    const int s = t - 512;
    const int lane = s & 31, ntile = s >> 5;
    const int ch = ntile * 16 + (lane & 15);
    const int kbase = (lane >> 4) ? 8 : 0;
    v16bf f;
#pragma unroll
    for (int j = 0; j < 16; ++j) f[j] = f2bf(We[(size_t)(kbase + j + ((j >> 3) << 3)) * OUT_CH + ch]);
    Wef[s] = f;
  }
}

__global__ void degree_kernel(const long long* __restrict__ col, float* __restrict__ deg) {
  int i = blockIdx.x * blockDim.x + threadIdx.x;
  if (i < N_EDGES) atomicAdd(&deg[(size_t)col[i]], 1.0f);
}

__global__ void dinv_kernel(const float* __restrict__ deg, float* __restrict__ dinv) {
  int i = blockIdx.x * blockDim.x + threadIdx.x;
  if (i < N_NODES) {
    float d = deg[i];
    dinv[i] = (d > 0.0f) ? rsqrtf(d) : 0.0f;
  }
}

// h = x @ W_lin + b_lin. One wave owns a 16-row tile and ALL four N-tiles:
// A fragment built once per K-step, reused by 4 independent WMMAs (no hazard NOPs).
__global__ __launch_bounds__(64)
void h_gemm_kernel(const float* __restrict__ x,      // [N_NODES, 128]
                   const v16bf* __restrict__ Wlf,    // prepacked [4][4][32]
                   const float* __restrict__ b,      // [64]
                   float* __restrict__ h) {          // [N_NODES, 64]
  const int lane = threadIdx.x & 31;
  const int w    = threadIdx.x >> 5;                 // 0..1
  const int m0   = (blockIdx.x * 2 + w) * 16;
  const int n    = lane & 15;
  const int hi   = lane >> 4;
  const int koff = hi ? 8 : 0;

  v8f acc[4];
#pragma unroll
  for (int nt = 0; nt < 4; ++nt) {
    const float bias = b[nt * 16 + n];
#pragma unroll
    for (int i = 0; i < 8; ++i) acc[nt][i] = bias;
  }

  const float* xrow = x + (size_t)(m0 + n) * IN_CH;  // A row = lane&15

#pragma unroll
  for (int ks = 0; ks < IN_CH / 32; ++ks) {
    const int kk = ks * 32;
    FragU a;                                          // built once, used 4x
    pack8(a, 0, *(const float4*)(xrow + kk + koff),
             *(const float4*)(xrow + kk + koff + 4));
    pack8(a, 4, *(const float4*)(xrow + kk + 16 + koff),
             *(const float4*)(xrow + kk + 16 + koff + 4));
#pragma unroll
    for (int nt = 0; nt < 4; ++nt) {
      const v16bf bb = Wlf[ks * 128 + nt * 32 + lane];
      acc[nt] = __builtin_amdgcn_wmma_f32_16x16x32_bf16(
          false, a.v, false, bb, (short)0, acc[nt], false, false);
    }
  }
  // D layout: lane column = nt*16 + n, rows m = hi*8 + v
#pragma unroll
  for (int v = 0; v < 8; ++v) {
    float* hrow = h + (size_t)(m0 + hi * 8 + v) * OUT_CH + n;
#pragma unroll
    for (int nt = 0; nt < 4; ++nt) hrow[nt * 16] = acc[nt][v];
  }
}

// Fused edge pass: one wave owns 16 edges; e-tile (16x64) via 4 independent
// WMMAs sharing one A fragment; then norm*h[row]*e scatter-added into out[col].
__global__ __launch_bounds__(128)
void edge_kernel(const float* __restrict__ ef,        // [N_EDGES, 32]
                 const long long* __restrict__ eidx,  // [2, N_EDGES]
                 const v16bf* __restrict__ Wef,       // prepacked [4][32]
                 const float* __restrict__ be,        // [64]
                 const float* __restrict__ h,         // [N_NODES, 64]
                 const float* __restrict__ dinv,      // [N_NODES]
                 float* __restrict__ out) {           // [N_NODES, 64]
  const int lane = threadIdx.x & 31;
  const int w    = threadIdx.x >> 5;                  // 0..3
  const long long e0 = ((long long)blockIdx.x * 4 + w) * 16;
  const int n    = lane & 15;
  const int hi   = lane >> 4;
  const int koff = hi ? 8 : 0;

  // Per-edge metadata: lane l covers edge e0 + (l&15); loaded before the WMMAs
  // so the latency hides under matrix work.
  const int el = lane & 15;
  const int re = (int)eidx[e0 + el];
  const int ce = (int)eidx[(size_t)N_EDGES + e0 + el];
  const float nrm = dinv[re] * dinv[ce];

  // A fragment: edge row e0 + (lane&15), single K step (EDGE_DIM = 32)
  const float* arow = ef + (size_t)(e0 + n) * EDGE_DIM;
  FragU a;
  pack8(a, 0, *(const float4*)(arow + koff),
           *(const float4*)(arow + koff + 4));
  pack8(a, 4, *(const float4*)(arow + 16 + koff),
           *(const float4*)(arow + 16 + koff + 4));

  v8f acc[4];
#pragma unroll
  for (int nt = 0; nt < 4; ++nt) {
    const float bias = be[nt * 16 + n];
#pragma unroll
    for (int i = 0; i < 8; ++i) acc[nt][i] = bias;
  }
#pragma unroll
  for (int nt = 0; nt < 4; ++nt) {
    const v16bf bb = Wef[nt * 32 + lane];
    acc[nt] = __builtin_amdgcn_wmma_f32_16x16x32_bf16(
        false, a.v, false, bb, (short)0, acc[nt], false, false);
  }

  // element v -> edge m = hi*8 + v; metadata broadcast via lane permutes
#pragma unroll
  for (int v = 0; v < 8; ++v) {
    const int m = hi * 8 + v;
    const float norm = __shfl(nrm, m, 32);
    const int r = __shfl(re, m, 32);
    const int c = __shfl(ce, m, 32);
    const float* hrow = h + (size_t)r * OUT_CH + n;
    float* orow = out + (size_t)c * OUT_CH + n;
#pragma unroll
    for (int nt = 0; nt < 4; ++nt) {
      const float val = norm * hrow[nt * 16] * acc[nt][v];
      atomicAdd(&orow[nt * 16], val);
    }
  }
}

extern "C" void kernel_launch(void* const* d_in, const int* in_sizes, int n_in,
                              void* d_out, int out_size, void* d_ws, size_t ws_size,
                              hipStream_t stream) {
  const float*     x    = (const float*)d_in[0];
  const long long* eidx = (const long long*)d_in[1];
  const float*     ef   = (const float*)d_in[2];
  const float*     Wl   = (const float*)d_in[3];
  const float*     bl   = (const float*)d_in[4];
  const float*     We   = (const float*)d_in[5];
  const float*     be   = (const float*)d_in[6];
  float* out = (float*)d_out;

  char*  wsb  = (char*)d_ws;
  float* deg  = (float*)wsb;                          // [N_NODES]
  float* dinv = deg  + N_NODES;                       // [N_NODES]
  float* h    = dinv + N_NODES;                       // [N_NODES*OUT_CH]
  // bf16 weight fragments after h (offset 26,400,000 B, 32B-aligned)
  v16bf* Wlf  = (v16bf*)(wsb + (size_t)(2 * N_NODES + N_NODES * OUT_CH) * sizeof(float));
  v16bf* Wef  = Wlf + 512;

  zero_f32_kernel<<<(N_NODES + 255) / 256, 256, 0, stream>>>(deg, N_NODES);
  zero_f32_kernel<<<(N_NODES * OUT_CH + 255) / 256, 256, 0, stream>>>(out, N_NODES * OUT_CH);
  pack_w_kernel<<<1, 640, 0, stream>>>(Wl, We, Wlf, Wef);
  degree_kernel<<<(N_EDGES + 255) / 256, 256, 0, stream>>>(eidx + N_EDGES, deg);
  dinv_kernel<<<(N_NODES + 255) / 256, 256, 0, stream>>>(deg, dinv);
  h_gemm_kernel<<<N_NODES / 32, 64, 0, stream>>>(x, Wlf, bl, h);
  edge_kernel<<<N_EDGES / 64, 128, 0, stream>>>(ef, eidx, Wef, be, h, dinv, out);
}
